// KnowCTIEmbedder_19250043421250
// MI455X (gfx1250) — matrix operand order repaired and checked by reading, not compiled, not run on hardware
//
#include <hip/hip_runtime.h>
#include <hip/hip_bf16.h>

// ---------------- problem constants ----------------
#define B_   16
#define L_   1024
#define D_   768
#define H_   4
#define C_   256
#define HC_  1024           // H*C
#define E_   8192
#define EN_  9216           // E + L self loops
#define M_   16384          // B*L rows
#define KCAT_ 1792          // D + HC
#define NEG_SLOPE_ 0.2f

#define LDS_COL_STRIDE 40   // 32 bf16 data + 8 pad (80 B) per staged B column

// ---------------- WMMA types ----------------
typedef __attribute__((ext_vector_type(16))) __bf16          bf16x16;
typedef __attribute__((ext_vector_type(16))) unsigned short  u16x16;
typedef __attribute__((ext_vector_type(8)))  unsigned short  u16x8;
typedef __attribute__((ext_vector_type(8)))  float           f32x8;

__device__ __forceinline__ unsigned short f2bf(float f) {
    unsigned u = __float_as_uint(f);
    u += 0x7FFFu + ((u >> 16) & 1u);          // round-to-nearest-even
    return (unsigned short)(u >> 16);
}

// build a 16-element bf16 fragment from two aligned 8x16-bit loads
__device__ __forceinline__ bf16x16 frag_from(const unsigned short* p0,
                                             const unsigned short* p1) {
    u16x8 lo = *(const u16x8*)p0;
    u16x8 hi = *(const u16x8*)p1;
    u16x16 v = __builtin_shufflevector(lo, hi, 0, 1, 2, 3, 4, 5, 6, 7,
                                               8, 9, 10, 11, 12, 13, 14, 15);
    return __builtin_bit_cast(bf16x16, v);
}

// ordered-uint mapping for float atomicMax
__device__ __forceinline__ unsigned fflip(float f) {
    unsigned u = __float_as_uint(f);
    return (u & 0x80000000u) ? ~u : (u | 0x80000000u);
}
__device__ __forceinline__ float funflip(unsigned u) {
    u = (u & 0x80000000u) ? (u & 0x7FFFFFFFu) : ~u;
    return __uint_as_float(u);
}

// ---------------- 0a: workspace init ----------------
__global__ void init_ws_kernel(float* __restrict__ G,
                               unsigned* __restrict__ mord,
                               float* __restrict__ denom) {
    size_t idx = (size_t)blockIdx.x * blockDim.x + threadIdx.x;
    if (idx < (size_t)M_ * HC_) G[idx] = 0.0f;
    if (idx < (size_t)M_ * H_) { mord[idx] = 0x007FFFFFu; /* fflip(-inf) */ denom[idx] = 0.0f; }
}

// ---------------- 0b: f32 -> bf16 pre-conversion (weights / activations) ----------------
__global__ void cvt_bf16_kernel(const float* __restrict__ src,
                                unsigned short* __restrict__ dst, int n) {
    int idx = blockIdx.x * 256 + threadIdx.x;
    if (idx < n) dst[idx] = f2bf(src[idx]);
}

// ---------------- 0c: (G + b_gat) -> bf16 after scatter ----------------
__global__ void cvt_g_bf16_kernel(const float* __restrict__ G,
                                  const float* __restrict__ bg,
                                  unsigned short* __restrict__ GBF) {
    size_t idx = (size_t)blockIdx.x * 256 + threadIdx.x;   // exactly M_*HC_ threads
    GBF[idx] = f2bf(G[idx] + bg[idx & (HC_ - 1)]);
}

// ---------------- 1: h = x @ W_gat^T  (bf16 WMMA, f32 acc, LDS-staged B) ----------------
// block: 256 thr = 8 waves; block tile 128 rows x 64 cols; grid = (HC/64, M/128)
__global__ __launch_bounds__(256) void gemm1_wmma(
    const unsigned short* __restrict__ XBF,   // [M_, D_] bf16
    const unsigned short* __restrict__ WgBF,  // [HC_, D_] bf16
    float* __restrict__ Hout)                 // [M_, HC_] f32
{
    __shared__ __align__(16) unsigned short ldsB[2][64 * LDS_COL_STRIDE];

    const int lane = threadIdx.x & 31;
    const int wave = threadIdx.x >> 5;
    const int r0 = blockIdx.y * 128 + wave * 16;
    const int c0 = blockIdx.x * 64;

    const int mA  = lane & 15;
    const int kbA = (lane >> 4) * 8;       // A: lanes 0-15 -> K{0..7,16..23}; 16-31 -> K{8..15,24..31}
    const int nB  = lane & 15;
    const int kbB = (lane >> 4) * 8;       // B: lanes 0-15 -> K{0..7,16..23} of col n; 16-31 -> K{8..15,24..31}

    // cooperative B staging: thread t loads 8 bf16 of column (t>>2), k-seg (t&3)*8
    const int scol = threadIdx.x >> 2;
    const int sseg = (threadIdx.x & 3) * 8;
    const unsigned short* srcB = WgBF + (size_t)(c0 + scol) * D_ + sseg;
    unsigned short* dstB0 = &ldsB[0][scol * LDS_COL_STRIDE + sseg];
    unsigned short* dstB1 = &ldsB[1][scol * LDS_COL_STRIDE + sseg];

    const unsigned short* arow = XBF + (size_t)(r0 + mA) * D_;

    f32x8 acc[4] = {};

    *(u16x8*)dstB0 = *(const u16x8*)srcB;   // prologue: stage chunk 0
    __syncthreads();

    const int NCH = D_ / 32;
    for (int c = 0; c < NCH; ++c) {
        const int k0 = c * 32;
        u16x8 nxt;
        if (c + 1 < NCH) nxt = *(const u16x8*)(srcB + (c + 1) * 32);

        bf16x16 a = frag_from(arow + k0 + kbA, arow + k0 + kbA + 16);
        const unsigned short* bbase = &ldsB[c & 1][0];
#pragma unroll
        for (int f = 0; f < 4; ++f) {
            const unsigned short* bp = bbase + (f * 16 + nB) * LDS_COL_STRIDE + kbB;
            bf16x16 b = frag_from(bp, bp + 16);
            acc[f] = __builtin_amdgcn_wmma_f32_16x16x32_bf16(
                false, a, false, b, (short)0, acc[f], false, false);
        }
        if (c + 1 < NCH) *(u16x8*)((c & 1) ? dstB0 : dstB1) = nxt;
        __syncthreads();
    }

    const int rOff = (lane >> 4) * 8;      // C layout: lanes 16-31 hold M=8..15
    const int cc   = lane & 15;
#pragma unroll
    for (int f = 0; f < 4; ++f)
#pragma unroll
        for (int r = 0; r < 8; ++r)
            Hout[(size_t)(r0 + rOff + r) * HC_ + (c0 + f * 16 + cc)] = acc[f][r];
}

// ---------------- 2: a_s / a_d per-node per-head dots ----------------
__global__ __launch_bounds__(256) void attn_dots_kernel(
    const float* __restrict__ Hfeat,
    const float* __restrict__ att_src, const float* __restrict__ att_dst,
    float* __restrict__ a_s, float* __restrict__ a_d)
{
    const int lane = threadIdx.x & 31;
    const int wave = threadIdx.x >> 5;
    const int n = blockIdx.x * 8 + wave;
    const float* hrow = Hfeat + (size_t)n * HC_;
#pragma unroll
    for (int hd = 0; hd < H_; ++hd) {
        float ps = 0.0f, pd = 0.0f;
#pragma unroll
        for (int j = 0; j < 8; ++j) {
            const int c = hd * C_ + lane + 32 * j;
            const float hv = hrow[c];
            ps += hv * att_src[c];
            pd += hv * att_dst[c];
        }
#pragma unroll
        for (int off = 16; off > 0; off >>= 1) {
            ps += __shfl_down(ps, off, 32);
            pd += __shfl_down(pd, off, 32);
        }
        if (lane == 0) { a_s[n * H_ + hd] = ps; a_d[n * H_ + hd] = pd; }
    }
}

__device__ __forceinline__ void edge_decode(const int* __restrict__ EI,
                                            int b, int i, int& src, int& dst) {
    if (i < E_) { src = EI[b * 2 * E_ + i]; dst = EI[b * 2 * E_ + E_ + i]; }
    else        { src = dst = i - E_; }   // self loop
}

// ---------------- 3: segment max of leaky-relu logits ----------------
__global__ void edge_max_kernel(const int* __restrict__ EI,
                                const float* __restrict__ a_s,
                                const float* __restrict__ a_d,
                                unsigned* __restrict__ mord)
{
    const int idx = blockIdx.x * 256 + threadIdx.x;
    if (idx >= B_ * EN_) return;
    const int b = idx / EN_, i = idx % EN_;
    int src, dst; edge_decode(EI, b, i, src, dst);
    const int ns = b * L_ + src, nd = b * L_ + dst;
#pragma unroll
    for (int hd = 0; hd < H_; ++hd) {
        float e = a_s[ns * H_ + hd] + a_d[nd * H_ + hd];
        e = (e > 0.0f) ? e : NEG_SLOPE_ * e;
        atomicMax(&mord[nd * H_ + hd], fflip(e));
    }
}

// ---------------- 4: segment sum of exp(e - m) ----------------
__global__ void edge_denom_kernel(const int* __restrict__ EI,
                                  const float* __restrict__ a_s,
                                  const float* __restrict__ a_d,
                                  const unsigned* __restrict__ mord,
                                  float* __restrict__ denom)
{
    const int idx = blockIdx.x * 256 + threadIdx.x;
    if (idx >= B_ * EN_) return;
    const int b = idx / EN_, i = idx % EN_;
    int src, dst; edge_decode(EI, b, i, src, dst);
    const int ns = b * L_ + src, nd = b * L_ + dst;
#pragma unroll
    for (int hd = 0; hd < H_; ++hd) {
        float e = a_s[ns * H_ + hd] + a_d[nd * H_ + hd];
        e = (e > 0.0f) ? e : NEG_SLOPE_ * e;
        const float m = funflip(mord[nd * H_ + hd]);
        atomicAdd(&denom[nd * H_ + hd], __expf(e - m));
    }
}

// ---------------- 5: gat[dst] += alpha * h[src]  (wave per edge*head) ----------------
__global__ __launch_bounds__(256) void gat_scatter_kernel(
    const int* __restrict__ EI,
    const float* __restrict__ a_s, const float* __restrict__ a_d,
    const unsigned* __restrict__ mord, const float* __restrict__ denom,
    const float* __restrict__ Hfeat, float* __restrict__ G)
{
    const int gtid = blockIdx.x * 256 + threadIdx.x;
    const int w = gtid >> 5;
    const int lane = gtid & 31;
    const int hd = w & 3;
    const int i  = (w >> 2) % EN_;
    const int b  = (w >> 2) / EN_;
    int src, dst; edge_decode(EI, b, i, src, dst);
    const int ns = b * L_ + src, nd = b * L_ + dst;
    float e = a_s[ns * H_ + hd] + a_d[nd * H_ + hd];
    e = (e > 0.0f) ? e : NEG_SLOPE_ * e;
    const float m = funflip(mord[nd * H_ + hd]);
    const float alpha = __expf(e - m) / (denom[nd * H_ + hd] + 1e-16f);
    const float* hsrc = Hfeat + (size_t)ns * HC_ + hd * C_;
    float* gdst       = G     + (size_t)nd * HC_ + hd * C_;
#pragma unroll
    for (int j = 0; j < 8; ++j) {
        const int c = lane + 32 * j;
        atomicAdd(&gdst[c], alpha * hsrc[c]);
    }
}

// ---------------- 6: out = concat(x, gat + b_gat) @ W_out^T + b_out ----------------
// same LDS-staged structure; A = [XBF | GBF] per row; grid = (D_/64, M/128)
__global__ __launch_bounds__(256) void gemm2_wmma(
    const unsigned short* __restrict__ XBF,   // [M_, D_] bf16
    const unsigned short* __restrict__ GBF,   // [M_, HC_] bf16 (includes +b_gat)
    const unsigned short* __restrict__ WoBF,  // [D_, KCAT_] bf16
    const float* __restrict__ bo,             // [D_]
    float* __restrict__ Out)                  // [M_, D_] f32
{
    __shared__ __align__(16) unsigned short ldsB[2][64 * LDS_COL_STRIDE];

    const int lane = threadIdx.x & 31;
    const int wave = threadIdx.x >> 5;
    const int r0 = blockIdx.y * 128 + wave * 16;
    const int c0 = blockIdx.x * 64;

    const int mA  = lane & 15;
    const int kbA = (lane >> 4) * 8;
    const int nB  = lane & 15;
    const int kbB = (lane >> 4) * 8;

    const int scol = threadIdx.x >> 2;
    const int sseg = (threadIdx.x & 3) * 8;
    const unsigned short* srcB = WoBF + (size_t)(c0 + scol) * KCAT_ + sseg;
    unsigned short* dstB0 = &ldsB[0][scol * LDS_COL_STRIDE + sseg];
    unsigned short* dstB1 = &ldsB[1][scol * LDS_COL_STRIDE + sseg];

    const unsigned short* xrow = XBF + (size_t)(r0 + mA) * D_;
    const unsigned short* grow = GBF + (size_t)(r0 + mA) * HC_;

    f32x8 acc[4] = {};

    *(u16x8*)dstB0 = *(const u16x8*)srcB;
    __syncthreads();

    const int NCH = KCAT_ / 32;
    for (int c = 0; c < NCH; ++c) {
        const int k0 = c * 32;
        u16x8 nxt;
        if (c + 1 < NCH) nxt = *(const u16x8*)(srcB + (c + 1) * 32);

        // A concat: each 8-element group is 8-aligned, never crosses the 768 seam
        const int ks0 = k0 + kbA, ks1 = k0 + kbA + 16;
        const unsigned short* p0 = (ks0 < D_) ? (xrow + ks0) : (grow + ks0 - D_);
        const unsigned short* p1 = (ks1 < D_) ? (xrow + ks1) : (grow + ks1 - D_);
        bf16x16 a = frag_from(p0, p1);

        const unsigned short* bbase = &ldsB[c & 1][0];
#pragma unroll
        for (int f = 0; f < 4; ++f) {
            const unsigned short* bp = bbase + (f * 16 + nB) * LDS_COL_STRIDE + kbB;
            bf16x16 b = frag_from(bp, bp + 16);
            acc[f] = __builtin_amdgcn_wmma_f32_16x16x32_bf16(
                false, a, false, b, (short)0, acc[f], false, false);
        }
        if (c + 1 < NCH) *(u16x8*)((c & 1) ? dstB0 : dstB1) = nxt;
        __syncthreads();
    }

    const int rOff = (lane >> 4) * 8;
    const int cc   = lane & 15;
#pragma unroll
    for (int f = 0; f < 4; ++f) {
        const int col = c0 + f * 16 + cc;
        const float bias = bo[col];
#pragma unroll
        for (int r = 0; r < 8; ++r)
            Out[(size_t)(r0 + rOff + r) * D_ + col] = acc[f][r] + bias;
    }
}

// ---------------- launcher ----------------
extern "C" void kernel_launch(void* const* d_in, const int* in_sizes, int n_in,
                              void* d_out, int out_size, void* d_ws, size_t ws_size,
                              hipStream_t stream) {
    const float* X    = (const float*)d_in[0];   // [B,L,D]
    const int*   EI   = (const int*)  d_in[1];   // [B,2,E]
    const float* Wg   = (const float*)d_in[2];   // [HC,D]
    const float* Asrc = (const float*)d_in[3];   // [H,C]
    const float* Adst = (const float*)d_in[4];   // [H,C]
    const float* bg   = (const float*)d_in[5];   // [HC]
    const float* Wo   = (const float*)d_in[6];   // [D, D+HC]
    const float* bo   = (const float*)d_in[7];   // [D]
    float* Out = (float*)d_out;

    // workspace layout
    float* ws   = (float*)d_ws;
    float* Hf   = ws;                                   // 16,777,216 f32   h = x@Wg^T
    float* G    = ws + (size_t)M_ * HC_;                // 16,777,216 f32   gat accumulator
    float* AS   = G  + (size_t)M_ * HC_;                // 65,536 f32
    float* AD   = AS + (size_t)M_ * H_;                 // 65,536 f32
    float* DEN  = AD + (size_t)M_ * H_;                 // 65,536 f32
    unsigned* MORD = (unsigned*)(DEN + (size_t)M_ * H_);// 65,536 u32
    unsigned short* WgBF = (unsigned short*)(MORD + (size_t)M_ * H_); // [HC,D]   bf16
    unsigned short* WoBF = WgBF + (size_t)HC_ * D_;                   // [D,KCAT] bf16
    unsigned short* XBF  = WoBF + (size_t)D_ * KCAT_;                 // [M,D]    bf16
    unsigned short* GBF  = XBF  + (size_t)M_ * D_;                    // [M,HC]   bf16

    // 0: init + bf16 pre-conversions
    init_ws_kernel<<<(M_ * HC_) / 256, 256, 0, stream>>>(G, MORD, DEN);
    cvt_bf16_kernel<<<(HC_ * D_) / 256, 256, 0, stream>>>(Wg, WgBF, HC_ * D_);
    cvt_bf16_kernel<<<(D_ * KCAT_) / 256, 256, 0, stream>>>(Wo, WoBF, D_ * KCAT_);
    cvt_bf16_kernel<<<(M_ * D_) / 256, 256, 0, stream>>>(X, XBF, M_ * D_);

    // 1: h = x @ W_gat^T
    gemm1_wmma<<<dim3(HC_ / 64, M_ / 128), 256, 0, stream>>>(XBF, WgBF, Hf);

    // 2: a_s / a_d
    attn_dots_kernel<<<M_ / 8, 256, 0, stream>>>(Hf, Asrc, Adst, AS, AD);

    // 3/4: softmax max + denom over dst neighborhoods
    const int edgeThreads = B_ * EN_;
    const int edgeBlocks  = (edgeThreads + 255) / 256;
    edge_max_kernel  <<<edgeBlocks, 256, 0, stream>>>(EI, AS, AD, MORD);
    edge_denom_kernel<<<edgeBlocks, 256, 0, stream>>>(EI, AS, AD, MORD, DEN);

    // 5: weighted scatter (one wave per edge*head)
    const long long waves = (long long)B_ * EN_ * H_;
    gat_scatter_kernel<<<(int)(waves * 32 / 256), 256, 0, stream>>>(
        EI, AS, AD, MORD, DEN, Hf, G);

    // 5b: (G + b_gat) -> bf16 for the output GEMM's A operand
    cvt_g_bf16_kernel<<<(M_ * HC_) / 256, 256, 0, stream>>>(G, bg, GBF);

    // 6: out = concat(x, gat+b_gat) @ W_out^T + b_out
    gemm2_wmma<<<dim3(D_ / 64, M_ / 128), 256, 0, stream>>>(XBF, GBF, WoBF, bo, Out);
}
